// APPNP_27642409517687
// MI455X (gfx1250) — matrix operand order repaired and checked by reading, not compiled, not run on hardware
//
#include <hip/hip_runtime.h>
#include <math.h>

// ---------------------------------------------------------------------------
// APPNP on MI455X (gfx1250, wave32):
//  - MLP via f16 WMMA (v_wmma_f32_16x16x32_f16), f32 accumulate
//    * block tile 256x64, wave tile 32x64 (8 wmma / K-step / wave)
//    * register-staged pipeline: next global tile fetched during wmma compute
//    * global_prefetch for k+2 slice
//  - propagation via atomic-free CSR pull (L2-resident working set)
// ---------------------------------------------------------------------------

typedef __attribute__((ext_vector_type(16))) _Float16 v16h;
typedef __attribute__((ext_vector_type(8)))  float    v8f;

#define NN 100000
#define NE 1600000

// ------------------------------- elementwise -------------------------------

__global__ void cvt_f32_f16(const float* __restrict__ in,
                            _Float16* __restrict__ out, long long n) {
  long long i = (long long)blockIdx.x * blockDim.x + threadIdx.x;
  long long stride = (long long)gridDim.x * blockDim.x;
  for (; i < n; i += stride) out[i] = (_Float16)in[i];
}

__global__ void zero_i32(int* __restrict__ p, int n) {
  int i = blockIdx.x * blockDim.x + threadIdx.x;
  int stride = gridDim.x * blockDim.x;
  for (; i < n; i += stride) p[i] = 0;
}

// ------------------------------ graph build --------------------------------

__global__ void deg_kernel(const int* __restrict__ dst, int* __restrict__ deg,
                           int nE) {
  int i = blockIdx.x * blockDim.x + threadIdx.x;
  if (i < nE) atomicAdd(&deg[dst[i]], 1);
}

__global__ void norm_kernel(const int* __restrict__ deg,
                            float* __restrict__ nrm, int n) {
  int i = blockIdx.x * blockDim.x + threadIdx.x;
  if (i < n) {
    int d = deg[i];
    nrm[i] = (d > 0) ? rsqrtf((float)d) : 1.0f;
  }
}

__global__ __launch_bounds__(1024)
void scan_kernel(const int* __restrict__ deg, int* __restrict__ offs,
                 int* __restrict__ cursor, int n) {
  __shared__ int sums[1024];
  int t = threadIdx.x;
  int chunk = (n + 1023) / 1024;
  int beg = t * chunk;
  int end = beg + chunk; if (end > n) end = n; if (beg > n) beg = n;
  int s = 0;
  for (int i = beg; i < end; ++i) s += deg[i];
  sums[t] = s;
  __syncthreads();
  for (int off = 1; off < 1024; off <<= 1) {
    int v = (t >= off) ? sums[t - off] : 0;
    __syncthreads();
    sums[t] += v;
    __syncthreads();
  }
  int run = (t == 0) ? 0 : sums[t - 1];
  for (int i = beg; i < end; ++i) {
    offs[i] = run; cursor[i] = run; run += deg[i];
  }
  if (t == 1023) offs[n] = sums[1023];
}

__global__ void fill_kernel(const int* __restrict__ src,
                            const int* __restrict__ dst,
                            int* __restrict__ cursor, int* __restrict__ csr,
                            int nE) {
  int i = blockIdx.x * blockDim.x + threadIdx.x;
  if (i < nE) {
    int p = atomicAdd(&cursor[dst[i]], 1);
    csr[p] = src[i];
  }
}

// -------------------------------- WMMA GEMM --------------------------------
// C[M,N] = act(A[M,K] @ B[K,N] + bias), A/B f16 row-major, acc f32.
// Block: 256 thr = 8 waves; block tile 256(M) x 64(N), K-step 32.
// Wave w -> rows [32w, 32w+32): 2 A-fragments x 4 B-fragments = 8 wmma.

template <bool RELU, bool OUT_F16>
__global__ __launch_bounds__(256)
void gemm_kernel(const _Float16* __restrict__ A, const _Float16* __restrict__ B,
                 const float* __restrict__ bias, void* __restrict__ Cout,
                 int M, int N, int K) {
  __shared__ _Float16 As[256][34];   // +2 halves pad: conflict-free frag reads
  __shared__ _Float16 Bs[64][34];    // stored transposed: Bs[n][k]

  const int gm = blockIdx.x * 256;
  const int gn = blockIdx.y * 64;
  const int wave = threadIdx.x >> 5;
  const int lane = threadIdx.x & 31;
  const int laneHi = lane >> 4;      // 0/1
  const int l15 = lane & 15;

  v8f c[2][4];
#pragma unroll
  for (int mf = 0; mf < 2; ++mf)
#pragma unroll
    for (int t = 0; t < 4; ++t)
#pragma unroll
      for (int r = 0; r < 8; ++r) c[mf][t][r] = 0.0f;

  // A-tile loader: one full 32-half row (64B) per thread
  const int ar = threadIdx.x;
  const int agrow = gm + ar;
  const bool aok = agrow < M;
  // B-tile loader: 8 threads per k-row, 8 halves each
  const int bk = threadIdx.x >> 3;
  const int bn = (threadIdx.x & 7) * 8;

  uint4 sa0, sa1, sa2, sa3, sb;

  // prologue: fetch K-step 0 into registers
  if (aok) {
    const uint4* p = (const uint4*)(A + (size_t)agrow * K);
    sa0 = p[0]; sa1 = p[1]; sa2 = p[2]; sa3 = p[3];
  } else {
    sa0 = make_uint4(0, 0, 0, 0); sa1 = sa0; sa2 = sa0; sa3 = sa0;
  }
  sb = *(const uint4*)(B + (size_t)bk * N + gn + bn);

  for (int k0 = 0; k0 < K; k0 += 32) {
    // ---- commit staged tile to LDS ----
    {
      unsigned* d = (unsigned*)&As[ar][0];  // 4B aligned (68*ar)
      d[0] = sa0.x;  d[1] = sa0.y;  d[2]  = sa0.z;  d[3]  = sa0.w;
      d[4] = sa1.x;  d[5] = sa1.y;  d[6]  = sa1.z;  d[7]  = sa1.w;
      d[8] = sa2.x;  d[9] = sa2.y;  d[10] = sa2.z;  d[11] = sa2.w;
      d[12] = sa3.x; d[13] = sa3.y; d[14] = sa3.z;  d[15] = sa3.w;
      union { uint4 u; _Float16 h[8]; } bu;
      bu.u = sb;
#pragma unroll
      for (int j = 0; j < 8; ++j) Bs[bn + j][bk] = bu.h[j];
    }
    __syncthreads();

    // ---- issue next-tile global fetch (overlaps with wmma below) ----
    const int kn = k0 + 32;
    if (kn < K) {
      if (aok) {
        const uint4* p = (const uint4*)(A + (size_t)agrow * K + kn);
        sa0 = p[0]; sa1 = p[1]; sa2 = p[2]; sa3 = p[3];
        if (kn + 32 < K)
          __builtin_prefetch(A + (size_t)agrow * K + kn + 32, 0, 1);
      }
      sb = *(const uint4*)(B + (size_t)(kn + bk) * N + gn + bn);
    }

    // ---- A fragments (ISA 16-bit A layout, wave32) ----
    v16h af[2];
#pragma unroll
    for (int mf = 0; mf < 2; ++mf) {
      const _Float16* arow = &As[wave * 32 + mf * 16 + l15][0];
#pragma unroll
      for (int v = 0; v < 8; ++v) {
        const int kk = ((v < 4) ? 2 * v : 16 + 2 * (v - 4)) + laneHi * 8;
        af[mf][2 * v]     = arow[kk];
        af[mf][2 * v + 1] = arow[kk + 1];
      }
    }
    // ---- 4 B fragments, each reused across both A fragments ----
#pragma unroll
    for (int t = 0; t < 4; ++t) {
      const _Float16* brow = &Bs[t * 16 + l15][0];
      v16h b;
#pragma unroll
      for (int v = 0; v < 8; ++v) {
        const int kk = ((v < 4) ? 2 * v : 16 + 2 * (v - 4)) + laneHi * 8;
        b[2 * v]     = brow[kk];
        b[2 * v + 1] = brow[kk + 1];
      }
      c[0][t] = __builtin_amdgcn_wmma_f32_16x16x32_f16(
          false, af[0], false, b, (short)0, c[0][t], false, false);
      c[1][t] = __builtin_amdgcn_wmma_f32_16x16x32_f16(
          false, af[1], false, b, (short)0, c[1][t], false, false);
    }
    __syncthreads();
  }

  // ---- epilogue: bias (+relu), store (f16 or f32) ----
#pragma unroll
  for (int mf = 0; mf < 2; ++mf) {
#pragma unroll
    for (int t = 0; t < 4; ++t) {
      const int col = gn + t * 16 + l15;
      const float bv = bias[col];
#pragma unroll
      for (int r = 0; r < 8; ++r) {
        const int row = gm + wave * 32 + mf * 16 + r + laneHi * 8;
        if (row < M) {
          float v = c[mf][t][r] + bv;
          if (RELU) v = fmaxf(v, 0.0f);
          if (OUT_F16)
            ((_Float16*)Cout)[(size_t)row * N + col] = (_Float16)v;
          else
            ((float*)Cout)[(size_t)row * N + col] = v;
        }
      }
    }
  }
}

// ------------------------------ propagation --------------------------------
// Pull-based: one wave per node, lane owns float4 slice of the 128-dim row.

__global__ __launch_bounds__(256)
void prop_kernel(const float4* __restrict__ hin4,
                 const float4* __restrict__ h04,
                 const float* __restrict__ nrm,
                 const int* __restrict__ offs,
                 const int* __restrict__ csr,
                 float4* __restrict__ hout4, int nNodes) {
  const int wave = threadIdx.x >> 5;
  const int lane = threadIdx.x & 31;
  const int node = blockIdx.x * 8 + wave;
  if (node >= nNodes) return;
  float4 acc = make_float4(0.f, 0.f, 0.f, 0.f);
  const int beg = offs[node], end = offs[node + 1];
  for (int j = beg; j < end; ++j) {
    const int s = csr[j];
    const float ns = nrm[s];
    const float4 v = hin4[(size_t)s * 32 + lane];
    acc.x += v.x * ns; acc.y += v.y * ns;
    acc.z += v.z * ns; acc.w += v.w * ns;
  }
  const float nn = nrm[node];
  const float4 h0v = h04[(size_t)node * 32 + lane];
  float4 o;
  o.x = 0.9f * (acc.x * nn) + 0.1f * h0v.x;
  o.y = 0.9f * (acc.y * nn) + 0.1f * h0v.y;
  o.z = 0.9f * (acc.z * nn) + 0.1f * h0v.z;
  o.w = 0.9f * (acc.w * nn) + 0.1f * h0v.w;
  hout4[(size_t)node * 32 + lane] = o;
}

// --------------------------- final linear + sigmoid ------------------------

__global__ __launch_bounds__(256)
void final_kernel(const float4* __restrict__ h4, const float* __restrict__ Wl,
                  const float* __restrict__ bl, float* __restrict__ out,
                  int nNodes) {
  const int wave = threadIdx.x >> 5;
  const int lane = threadIdx.x & 31;
  const int node = blockIdx.x * 8 + wave;
  if (node >= nNodes) return;
  const float4 h = h4[(size_t)node * 32 + lane];
  const float4 w = ((const float4*)Wl)[lane];
  float p = h.x * w.x + h.y * w.y + h.z * w.z + h.w * w.w;
#pragma unroll
  for (int off = 16; off > 0; off >>= 1) p += __shfl_xor(p, off, 32);
  if (lane == 0) out[node] = 1.0f / (1.0f + expf(-(p + bl[0])));
}

// --------------------------------- launch ----------------------------------

extern "C" void kernel_launch(void* const* d_in, const int* in_sizes, int n_in,
                              void* d_out, int out_size, void* d_ws,
                              size_t ws_size, hipStream_t stream) {
  (void)in_sizes; (void)n_in; (void)out_size; (void)ws_size;
  const float* feats = (const float*)d_in[0];
  const int*   src   = (const int*)d_in[1];
  const int*   dst   = (const int*)d_in[2];
  const float* W1 = (const float*)d_in[3];
  const float* b1 = (const float*)d_in[4];
  const float* W2 = (const float*)d_in[5];
  const float* b2 = (const float*)d_in[6];
  const float* W3 = (const float*)d_in[7];
  const float* b3 = (const float*)d_in[8];
  const float* Wl = (const float*)d_in[9];
  const float* bl = (const float*)d_in[10];
  float* out = (float*)d_out;

  char* ws = (char*)d_ws;
  // region 0: features-f16 (102.4MB) -> reused as h0 + hA after GEMM1
  _Float16* A16 = (_Float16*)(ws + 0);
  float*    h0  = (float*)(ws + 0);
  float*    hA  = (float*)(ws + 51200000);
  // region 1: h1-f16 (102.4MB) -> reused as hB + graph after GEMM2
  _Float16* h1     = (_Float16*)(ws + 102400000);
  float*    hB     = (float*)(ws + 102400000);
  float*    nrm    = (float*)(ws + 153600000);
  int*      deg    = (int*)(ws + 154000000);
  int*      offs   = (int*)(ws + 154400000);   // NN+1 ints
  int*      cursor = (int*)(ws + 154800128);
  int*      csr    = (int*)(ws + 155200128);   // NE ints
  // region 2: h2-f16 (51.2MB)
  _Float16* h2  = (_Float16*)(ws + 204800000);
  // region 3: weights f16
  _Float16* W1h = (_Float16*)(ws + 256000000);
  _Float16* W2h = (_Float16*)(ws + 256524288);
  _Float16* W3h = (_Float16*)(ws + 256786432);

  // 1) convert activations/weights to f16 for the WMMA path
  cvt_f32_f16<<<4096, 256, 0, stream>>>(feats, A16, (long long)NN * 512);
  cvt_f32_f16<<<256, 256, 0, stream>>>(W1, W1h, 512LL * 512);
  cvt_f32_f16<<<128, 256, 0, stream>>>(W2, W2h, 512LL * 256);
  cvt_f32_f16<<<32, 256, 0, stream>>>(W3, W3h, 256LL * 128);

  // 2) MLP GEMMs (WMMA); M blocks = ceil(100000/256) = 391
  gemm_kernel<true, true><<<dim3(391, 8), 256, 0, stream>>>(
      A16, W1h, b1, (void*)h1, NN, 512, 512);
  gemm_kernel<true, true><<<dim3(391, 4), 256, 0, stream>>>(
      h1, W2h, b2, (void*)h2, NN, 256, 512);

  // 3) graph build (after GEMM2: its region aliases h1)
  zero_i32<<<256, 256, 0, stream>>>(deg, NN);
  deg_kernel<<<6250, 256, 0, stream>>>(dst, deg, NE);
  norm_kernel<<<391, 256, 0, stream>>>(deg, nrm, NN);
  scan_kernel<<<1, 1024, 0, stream>>>(deg, offs, cursor, NN);
  fill_kernel<<<6250, 256, 0, stream>>>(src, dst, cursor, csr, NE);

  // 4) h0 = h2 @ W3 + b3  (f32 out, no relu)
  gemm_kernel<false, false><<<dim3(391, 2), 256, 0, stream>>>(
      h2, W3h, b3, (void*)h0, NN, 128, 256);

  // 5) 10 APPNP iterations, ping-pong hA/hB (ends in hB)
  const float* hin = h0;
  float* bufs[2] = {hA, hB};
  for (int it = 0; it < 10; ++it) {
    float* hout = bufs[it & 1];
    prop_kernel<<<12500, 256, 0, stream>>>(
        (const float4*)hin, (const float4*)h0, nrm, offs, csr,
        (float4*)hout, NN);
    hin = hout;
  }

  // 6) sigmoid(h @ Wl + bl)
  final_kernel<<<12500, 256, 0, stream>>>((const float4*)hin, Wl, bl, out, NN);
}
